// Block_60129542389
// MI455X (gfx1250) — matrix operand order repaired and checked by reading, not compiled, not run on hardware
//
#include <hip/hip_runtime.h>

// ---------------------------------------------------------------------------
// Problem constants (match the reference)
// ---------------------------------------------------------------------------
#define BQ 8
#define TQ 2048
#define CQ 768
#define HQ 3072
#define NTOK (BQ * TQ)   // 16384 tokens

typedef __attribute__((ext_vector_type(16))) __bf16        v16bf;
typedef __attribute__((ext_vector_type(8)))  float         v8f;
typedef __attribute__((ext_vector_type(8)))  unsigned int  v8u;

// ---------------------------------------------------------------------------
// bf16 helpers (round-to-nearest-even)
// ---------------------------------------------------------------------------
__device__ __forceinline__ unsigned short f2bf(float f) {
    unsigned int u = __float_as_uint(f);
    unsigned int r = u + 0x7FFFu + ((u >> 16) & 1u);
    return (unsigned short)(r >> 16);
}

__global__ __launch_bounds__(256) void cvt_bf16(const float* __restrict__ in,
                                                unsigned short* __restrict__ out,
                                                int n) {
    int i = blockIdx.x * blockDim.x + threadIdx.x;
    if (i < n) out[i] = f2bf(in[i]);
}

// ---------------------------------------------------------------------------
// LayerNorm over C per token; writes bf16 activations for the WMMA GEMMs.
// ---------------------------------------------------------------------------
__global__ __launch_bounds__(256) void ln_bf16(const float* __restrict__ x,
                                               const float* __restrict__ w,
                                               const float* __restrict__ b,
                                               unsigned short* __restrict__ out,
                                               int Cc) {
    __shared__ float s1[256];
    __shared__ float s2[256];
    const int token = blockIdx.x;
    const float* xr = x + (size_t)token * Cc;
    float a = 0.f, q = 0.f;
    for (int c = threadIdx.x; c < Cc; c += 256) {
        float v = xr[c];
        a += v; q += v * v;
    }
    s1[threadIdx.x] = a; s2[threadIdx.x] = q;
    __syncthreads();
    for (int s = 128; s > 0; s >>= 1) {
        if ((int)threadIdx.x < s) {
            s1[threadIdx.x] += s1[threadIdx.x + s];
            s2[threadIdx.x] += s2[threadIdx.x + s];
        }
        __syncthreads();
    }
    const float mu   = s1[0] / (float)Cc;
    const float var  = s2[0] / (float)Cc - mu * mu;
    const float rstd = rsqrtf(var + 1e-5f);
    unsigned short* orow = out + (size_t)token * Cc;
    for (int c = threadIdx.x; c < Cc; c += 256) {
        float v = (xr[c] - mu) * rstd * w[c] + b[c];
        orow[c] = f2bf(v);
    }
}

// ---------------------------------------------------------------------------
// WMMA bf16 GEMM:  Out[i, j] = sum_k A[i, k] * W[j, k]      (i: token, j: feat)
// Block tile 128 tok x 128 feat; K-chunk 32; 8 waves (4 tok x 2 feat groups);
// each wave: 2 A-frags x 4 B-frags = 8 v_wmma per chunk.
// Double-buffered LDS staged with GLOBAL_LOAD_ASYNC_TO_LDS_B128 (ASYNCcnt):
// next chunk's loads are issued right after the barrier, overlapping WMMA.
// Compile-time epilogues (EPI):
//   0: outF = acc
//   1: outF = sigmoid(acc)
//   2: outF = aux0 + aux1 * acc          (residual + gate)
//   3: outH = bf16(relu(acc)^2)
// ---------------------------------------------------------------------------
#define APITCH 40
#define BPITCH 40
#define ABUF (128 * APITCH)   // halfs per A buffer
#define BBUF (128 * BPITCH)   // halfs per B buffer

__device__ __forceinline__ void async_b128(unsigned lds_off,
                                           const unsigned short* gaddr) {
    asm volatile("global_load_async_to_lds_b128 %0, %1, off"
                 :: "v"(lds_off), "v"(gaddr)
                 : "memory");
}

__device__ __forceinline__ void wait_async0() {
    asm volatile("s_wait_asynccnt 0" ::: "memory");
}

__device__ __forceinline__ v16bf frag_a(const unsigned short* __restrict__ base,
                                        int lane) {
    // 16-bit A-matrix 16x32 layout (ISA 7.12.2)
    const int row  = lane & 15;
    const int kofs = (lane >> 4) ? 8 : 0;
    const unsigned short* p = base + row * APITCH;
    v8u u;
#pragma unroll
    for (int i = 0; i < 8; ++i) {
        int k = ((i < 4) ? (2 * i) : (16 + 2 * (i - 4))) + kofs;
        u[i] = *(const unsigned int*)(p + k);
    }
    return __builtin_bit_cast(v16bf, u);
}

__device__ __forceinline__ v16bf frag_b(const unsigned short* __restrict__ base,
                                        int lane) {
    // 16-bit B-matrix 32x16: same K order as A (lanes 0-15 K=0..7,16..23; lanes 16-31 K=8..15,24..31)
    const int col  = lane & 15;
    const int kofs = (lane >> 4) ? 8 : 0;
    const unsigned short* p = base + col * BPITCH;
    v8u u;
#pragma unroll
    for (int i = 0; i < 8; ++i) {
        int k = ((i < 4) ? (2 * i) : (16 + 2 * (i - 4))) + kofs;
        u[i] = *(const unsigned int*)(p + k);
    }
    return __builtin_bit_cast(v16bf, u);
}

__device__ __forceinline__ v8f wmma_bf16(v16bf a, v16bf b, v8f c) {
    return __builtin_amdgcn_wmma_f32_16x16x32_bf16(
        /*neg_a=*/false, a, /*neg_b=*/false, b,
        /*c_mod=*/(short)0, c, /*reuse_a=*/false, /*reuse_b=*/false);
}

__device__ __forceinline__ void mma_chunk(const unsigned short* __restrict__ Asb,
                                          const unsigned short* __restrict__ Bsb,
                                          int waveTok, int waveFeat, int lane,
                                          v8f (&acc)[2][4]) {
    v16bf a0 = frag_a(Asb + (waveTok + 0)  * APITCH, lane);
    v16bf a1 = frag_a(Asb + (waveTok + 16) * APITCH, lane);
#pragma unroll
    for (int fi = 0; fi < 4; ++fi) {
        v16bf b = frag_b(Bsb + (waveFeat + fi * 16) * BPITCH, lane);
        acc[0][fi] = wmma_bf16(a0, b, acc[0][fi]);
        acc[1][fi] = wmma_bf16(a1, b, acc[1][fi]);
    }
}

template <int EPI>
__global__ __launch_bounds__(256) void gemm_bf16_wmma(
    const unsigned short* __restrict__ A,
    const unsigned short* __restrict__ W,
    int K, int Mout,
    float* __restrict__ outF, unsigned short* __restrict__ outH,
    const float* __restrict__ aux0, const float* __restrict__ aux1) {
    __shared__ unsigned short As[2 * ABUF];
    __shared__ unsigned short Bs[2 * BBUF];

    const int tid      = threadIdx.x;
    const int lane     = tid & 31;
    const int wid      = tid >> 5;
    const int tokBase  = blockIdx.y * 128;
    const int featBase = blockIdx.x * 128;
    const int waveTok  = (wid >> 1) * 32;   // 0,32,64,96
    const int waveFeat = (wid & 1) * 64;    // 0,64

    v8f acc[2][4];
#pragma unroll
    for (int i = 0; i < 2; ++i)
#pragma unroll
        for (int j = 0; j < 4; ++j)
            acc[i][j] = (v8f){0.f, 0.f, 0.f, 0.f, 0.f, 0.f, 0.f, 0.f};

    // Staging map: 256 threads cover a 128 x 32-half tile in two 64-row passes,
    // 16 bytes (8 halfs) per thread per pass.
    const int srow = tid >> 2;          // 0..63
    const int skol = (tid & 3) * 8;     // 0,8,16,24

    const unsigned short* aG0 = A + (size_t)(tokBase + srow)       * K + skol;
    const unsigned short* aG1 = A + (size_t)(tokBase + srow + 64)  * K + skol;
    const unsigned short* bG0 = W + (size_t)(featBase + srow)      * K + skol;
    const unsigned short* bG1 = W + (size_t)(featBase + srow + 64) * K + skol;

    const unsigned lA0 = (unsigned)(uintptr_t)&As[(srow)      * APITCH + skol];
    const unsigned lA1 = (unsigned)(uintptr_t)&As[(srow + 64) * APITCH + skol];
    const unsigned lB0 = (unsigned)(uintptr_t)&Bs[(srow)      * BPITCH + skol];
    const unsigned lB1 = (unsigned)(uintptr_t)&Bs[(srow + 64) * BPITCH + skol];
    const unsigned aStride = ABUF * 2;  // bytes between A buffers
    const unsigned bStride = BBUF * 2;  // bytes between B buffers

    const int nIter = K >> 5;           // 24 or 96 — always even here

    // Prologue: stage chunk 0 into buffer 0.
    async_b128(lA0, aG0); async_b128(lA1, aG1);
    async_b128(lB0, bG0); async_b128(lB1, bG1);
    aG0 += 32; aG1 += 32; bG0 += 32; bG1 += 32;

    for (int it = 0; it < nIter; it += 2) {
        // ---- even chunk: read buf0, prefetch buf1 ----
        wait_async0();
        __syncthreads();
        if (it + 1 < nIter) {
            async_b128(lA0 + aStride, aG0); async_b128(lA1 + aStride, aG1);
            async_b128(lB0 + bStride, bG0); async_b128(lB1 + bStride, bG1);
            aG0 += 32; aG1 += 32; bG0 += 32; bG1 += 32;
        }
        mma_chunk(&As[0], &Bs[0], waveTok, waveFeat, lane, acc);

        // ---- odd chunk: read buf1, prefetch buf0 ----
        wait_async0();
        __syncthreads();
        if (it + 2 < nIter) {
            async_b128(lA0, aG0); async_b128(lA1, aG1);
            async_b128(lB0, bG0); async_b128(lB1, bG1);
            aG0 += 32; aG1 += 32; bG0 += 32; bG1 += 32;
        }
        mma_chunk(&As[ABUF], &Bs[BBUF], waveTok, waveFeat, lane, acc);
    }

    // Epilogue: C/D 16x16 f32 layout — lane&15 = N (feat), vgpr r + 8*(lane>=16) = M (token)
    const int col0 = featBase + waveFeat + (lane & 15);
    const int row0 = tokBase + waveTok + ((lane >> 4) ? 8 : 0);
#pragma unroll
    for (int ti = 0; ti < 2; ++ti) {
#pragma unroll
        for (int fi = 0; fi < 4; ++fi) {
            const int col  = col0 + fi * 16;
            const int rowb = row0 + ti * 16;
#pragma unroll
            for (int r = 0; r < 8; ++r) {
                const int token = rowb + r;
                const size_t idx = (size_t)token * Mout + col;
                const float v = acc[ti][fi][r];
                if constexpr (EPI == 0) {
                    outF[idx] = v;
                } else if constexpr (EPI == 1) {
                    outF[idx] = 1.f / (1.f + __expf(-v));
                } else if constexpr (EPI == 2) {
                    outF[idx] = aux0[idx] + aux1[idx] * v;
                } else {
                    float t = v > 0.f ? v : 0.f;
                    outH[idx] = f2bf(t * t);
                }
            }
        }
    }
}

// ---------------------------------------------------------------------------
// Bidirectional WKV. One thread per (b, c) channel; sequential over T.
// ---------------------------------------------------------------------------
__global__ __launch_bounds__(256) void wkv_fwd(
    const float* __restrict__ k, const float* __restrict__ v,
    const float* __restrict__ decay,
    float* __restrict__ af, float* __restrict__ bfv, float* __restrict__ mf) {
    const int idx = blockIdx.x * blockDim.x + threadIdx.x;
    if (idx >= BQ * CQ) return;
    const int b = idx / CQ;
    const int c = idx % CQ;
    const float w = decay[c] / (float)TQ;
    float a = 0.f, bb = 0.f, m = -1e38f;
    const size_t off = (size_t)b * TQ * CQ + c;
    for (int t = 0; t < TQ; ++t) {
        const size_t p = off + (size_t)t * CQ;
        af[p] = a; bfv[p] = bb; mf[p] = m;
        const float kt = k[p], vt = v[p];
        const float mm = fmaxf(m + w, kt);
        const float e1 = __expf(m + w - mm);
        const float e2 = __expf(kt - mm);
        a = e1 * a + e2 * vt;
        bb = e1 * bb + e2;
        m = mm;
    }
}

__global__ __launch_bounds__(256) void wkv_bwd_combine(
    const float* __restrict__ k, const float* __restrict__ v,
    const float* __restrict__ af, const float* __restrict__ bfv,
    const float* __restrict__ mf,
    const float* __restrict__ decay, const float* __restrict__ first,
    unsigned short* __restrict__ ybf) {
    const int idx = blockIdx.x * blockDim.x + threadIdx.x;
    if (idx >= BQ * CQ) return;
    const int b = idx / CQ;
    const int c = idx % CQ;
    const float w = decay[c] / (float)TQ;
    const float u = first[c] / (float)TQ;
    float a = 0.f, bb = 0.f, m = -1e38f;   // exclusive backward carry
    const size_t off = (size_t)b * TQ * CQ + c;
    for (int t = TQ - 1; t >= 0; --t) {
        const size_t p = off + (size_t)t * CQ;
        const float kt = k[p], vt = v[p];
        const float Af = af[p], Bf = bfv[p], Mf = mf[p];
        const float kk = kt + u;
        const float M  = fmaxf(fmaxf(Mf, m), kk);
        const float ef = __expf(Mf - M);
        const float eb = __expf(m - M);
        const float ec = __expf(kk - M);
        const float num = ef * Af + eb * a + ec * vt;
        const float den = ef * Bf + eb * bb + ec;
        ybf[p] = f2bf(num / den);
        const float mm = fmaxf(m + w, kt);
        const float e1 = __expf(m + w - mm);
        const float e2 = __expf(kt - mm);
        a = e1 * a + e2 * vt;
        bb = e1 * bb + e2;
        m = mm;
    }
}

// ---------------------------------------------------------------------------
// Launch
// ---------------------------------------------------------------------------
extern "C" void kernel_launch(void* const* d_in, const int* in_sizes, int n_in,
                              void* d_out, int out_size, void* d_ws, size_t ws_size,
                              hipStream_t stream) {
    (void)in_sizes; (void)n_in; (void)out_size; (void)ws_size;

    const float* x     = (const float*)d_in[0];
    const float* ln1w  = (const float*)d_in[1];
    const float* ln1b  = (const float*)d_in[2];
    const float* ln2w  = (const float*)d_in[3];
    const float* ln2b  = (const float*)d_in[4];
    const float* decay = (const float*)d_in[5];
    const float* first = (const float*)d_in[6];
    const float* Wk_a  = (const float*)d_in[7];
    const float* Wv_a  = (const float*)d_in[8];
    const float* Wr_a  = (const float*)d_in[9];
    const float* Wo_a  = (const float*)d_in[10];
    const float* Wk_f  = (const float*)d_in[11];
    const float* Wv_f  = (const float*)d_in[12];
    const float* Wr_f  = (const float*)d_in[13];

    // ---- workspace layout (256B aligned bumps) ----
    char* p = (char*)d_ws;
    auto alloc = [&](size_t bytes) -> char* {
        char* r = p;
        p += (bytes + 255) & ~(size_t)255;
        return r;
    };
    unsigned short* wk  = (unsigned short*)alloc((size_t)CQ * CQ * 2);
    unsigned short* wv  = (unsigned short*)alloc((size_t)CQ * CQ * 2);
    unsigned short* wr  = (unsigned short*)alloc((size_t)CQ * CQ * 2);
    unsigned short* wo  = (unsigned short*)alloc((size_t)CQ * CQ * 2);
    unsigned short* wkf = (unsigned short*)alloc((size_t)HQ * CQ * 2);
    unsigned short* wvf = (unsigned short*)alloc((size_t)CQ * HQ * 2);
    unsigned short* wrf = (unsigned short*)alloc((size_t)CQ * CQ * 2);

    unsigned short* hbf = (unsigned short*)alloc((size_t)NTOK * CQ * 2); // h / later y
    float* kbuf = (float*)alloc((size_t)NTOK * CQ * 4);                  // k / later kf lo
    float* vbuf = (float*)alloc((size_t)NTOK * CQ * 4);                  // v / later kf hi
    float* sr   = (float*)alloc((size_t)NTOK * CQ * 4);                  // sr / later rf
    float* af   = (float*)alloc((size_t)NTOK * CQ * 4);                  // af / later x1
    float* bfv  = (float*)alloc((size_t)NTOK * CQ * 4);                  // bf / later h2(bf16)
    float* mf   = (float*)alloc((size_t)NTOK * CQ * 4);                  // mf

    unsigned short* ybf  = hbf;                    // y overwrites h
    float*          x1   = af;                     // x1 overwrites af
    unsigned short* h2bf = (unsigned short*)bfv;   // h2 overwrites bf
    unsigned short* kfbf = (unsigned short*)kbuf;  // kf (N x H bf16) == k||v region
    float*          rf   = sr;                     // rf overwrites sr
    float*          xout = (float*)d_out;

    // ---- 1) pack weights to bf16 ----
    auto cvt = [&](const float* in, unsigned short* out, int n) {
        cvt_bf16<<<(n + 255) / 256, 256, 0, stream>>>(in, out, n);
    };
    cvt(Wk_a, wk,  CQ * CQ);
    cvt(Wv_a, wv,  CQ * CQ);
    cvt(Wr_a, wr,  CQ * CQ);
    cvt(Wo_a, wo,  CQ * CQ);
    cvt(Wk_f, wkf, HQ * CQ);
    cvt(Wv_f, wvf, CQ * HQ);
    cvt(Wr_f, wrf, CQ * CQ);

    const dim3 blk(256);
    const dim3 gC(CQ / 128, NTOK / 128);  // (6, 128)
    const dim3 gH(HQ / 128, NTOK / 128);  // (24, 128)
    const int chanBlocks = (BQ * CQ + 255) / 256;

    // ---- 2) LN1 ----
    ln_bf16<<<NTOK, blk, 0, stream>>>(x, ln1w, ln1b, hbf, CQ);

    // ---- 3-5) k, v, sigmoid(r) ----
    gemm_bf16_wmma<0><<<gC, blk, 0, stream>>>(hbf, wk, CQ, CQ, kbuf, nullptr, nullptr, nullptr);
    gemm_bf16_wmma<0><<<gC, blk, 0, stream>>>(hbf, wv, CQ, CQ, vbuf, nullptr, nullptr, nullptr);
    gemm_bf16_wmma<1><<<gC, blk, 0, stream>>>(hbf, wr, CQ, CQ, sr,   nullptr, nullptr, nullptr);

    // ---- 6-7) bidirectional WKV ----
    wkv_fwd<<<chanBlocks, blk, 0, stream>>>(kbuf, vbuf, decay, af, bfv, mf);
    wkv_bwd_combine<<<chanBlocks, blk, 0, stream>>>(kbuf, vbuf, af, bfv, mf,
                                                    decay, first, ybf);

    // ---- 8) x1 = x + sr * (y @ Wo^T) ----
    gemm_bf16_wmma<2><<<gC, blk, 0, stream>>>(ybf, wo, CQ, CQ, x1, nullptr, x, sr);

    // ---- 9) LN2 ----
    ln_bf16<<<NTOK, blk, 0, stream>>>(x1, ln2w, ln2b, h2bf, CQ);

    // ---- 10) kf = relu(h2 @ Wk_f^T)^2  (bf16) ----
    gemm_bf16_wmma<3><<<gH, blk, 0, stream>>>(h2bf, wkf, CQ, HQ, nullptr, kfbf, nullptr, nullptr);

    // ---- 11) rf = sigmoid(h2 @ Wr_f^T) ----
    gemm_bf16_wmma<1><<<gC, blk, 0, stream>>>(h2bf, wrf, CQ, CQ, rf, nullptr, nullptr, nullptr);

    // ---- 12) out = x1 + rf * (kf @ Wv_f^T) ----
    gemm_bf16_wmma<2><<<gC, blk, 0, stream>>>(kfbf, wvf, HQ, CQ, xout, nullptr, x1, rf);
}
